// Attention_85375359910650
// MI455X (gfx1250) — matrix operand order repaired and checked by reading, not compiled, not run on hardware
//
#include <hip/hip_runtime.h>
#include <hip/hip_bf16.h>

typedef _Float16 f16;
typedef __attribute__((ext_vector_type(16))) _Float16 v16h;
typedef __attribute__((ext_vector_type(8)))  float    v8f;

// Problem constants
#define BB     64
#define TT     500
#define INC    256
#define NE     512
#define NH     8
#define HD     64        // head dim
#define QKVW   1536      // 3*NE
#define MT     2000      // B*T/16 == 32000/16 exactly
#define STILES 32        // 512/16 padded key tiles

// ---- WMMA lane-layout helpers (ISA 7.12.2, wave32) ----
// A-matrix (16-bit, 16x32): lane l holds row m=l&15; element e -> k:
__device__ __forceinline__ int amap_k(int e, int lane) {
    return (e & 7) + ((e >> 3) << 4) + ((lane >> 4) << 3);
}
// B-matrix (16-bit, 32x16): lane l holds col n=l&15; element e -> k:
__device__ __forceinline__ int bmap_k(int e, int lane) {
    return (e & 7) + ((e >> 3) << 4) + ((lane >> 4) << 3);
}
// C/D (f32 16x16): element i at lane l -> row = i + 8*(l>=16), col = l&15.

// =================== Kernel 0: one-shot f32 -> f16 conversion ===================
__global__ __launch_bounds__(256)
void cvt_kernel(const float* __restrict__ src, f16* __restrict__ dst, long n)
{
    long i = ((long)blockIdx.x * 256 + threadIdx.x) * 4;
    if (i < n) {
        float4 v = *(const float4*)(src + i);
        dst[i + 0] = (f16)v.x;
        dst[i + 1] = (f16)v.y;
        dst[i + 2] = (f16)v.z;
        dst[i + 3] = (f16)v.w;
    }
}

// =================== Kernel 1: QKV projection ===================
// qkv[r][n] = sum_k x[r][k] * w_attn[n][k] + b_attn[n]   (f16 out)
// One wave per 16x64 strip: A fragment feeds 4 WMMAs per k-step.
__global__ __launch_bounds__(128)
void qkv_proj_kernel(const f16* __restrict__ xh,
                     const f16* __restrict__ wh,
                     const float* __restrict__ bias,
                     f16* __restrict__ qkv)
{
    const int lane = threadIdx.x & 31;
    const int wid  = threadIdx.x >> 5;
    const int mt   = blockIdx.x * 4 + wid;    // grid.x = 500
    const int ntb  = blockIdx.y;              // grid.y = 24 (4 N-tiles each)
    const long row = (long)mt * 16 + (lane & 15);
    const int  n   = lane & 15;

    v8f c[4];
#pragma unroll
    for (int j = 0; j < 4; ++j) c[j] = (v8f){};

#pragma unroll
    for (int kb = 0; kb < INC; kb += 32) {
        v16h a;
#pragma unroll
        for (int e = 0; e < 16; ++e)
            a[e] = xh[row * INC + kb + amap_k(e, lane)];
#pragma unroll
        for (int j = 0; j < 4; ++j) {
            const long ncol = (long)(ntb * 4 + j) * 16 + n;
            v16h b;
#pragma unroll
            for (int e = 0; e < 16; ++e)
                b[e] = wh[ncol * INC + kb + bmap_k(e, lane)];
            c[j] = __builtin_amdgcn_wmma_f32_16x16x32_f16(false, a, false, b,
                                                          (short)0, c[j], false, false);
        }
    }
#pragma unroll
    for (int j = 0; j < 4; ++j) {
        const long ncol = (long)(ntb * 4 + j) * 16 + n;
        const float bn  = bias[ncol];
#pragma unroll
        for (int i = 0; i < 8; ++i) {
            int  mm = i + ((lane & 16) >> 1);
            long r  = (long)mt * 16 + mm;
            qkv[r * QKVW + ncol] = (f16)(c[j][i] + bn);
        }
    }
}

// =================== Kernel 2: fused flash attention per (b,h) ===================
// LDS: K[512][64] f16 (64KB) + V[512][64] f16 (64KB) + P scratch 8*16*32 f16 (8KB)
__global__ __launch_bounds__(256)
void attn_kernel(const f16* __restrict__ qkv, f16* __restrict__ y)
{
    extern __shared__ char smem_raw[];
    f16* Ks = (f16*)smem_raw;        // [512][64]
    f16* Vs = Ks + 512 * HD;         // [512][64]
    f16* Ps = Vs + 512 * HD;         // [8 waves][16][32]

    const int tid  = threadIdx.x;
    const int lane = tid & 31;
    const int wid  = tid >> 5;
    const int b    = blockIdx.x >> 3;
    const int h    = blockIdx.x & 7;

    const long base = (long)b * TT * QKVW;
    const f16* Qg = qkv + base +            h * HD;
    const f16* Kg = qkv + base +  NE      + h * HD;
    const f16* Vg = qkv + base +  2 * NE  + h * HD;

    // Stage K,V into LDS, zero-padded to 512 rows; 8 contiguous halves/thread/iter.
    for (int cidx = tid; cidx < 512 * HD / 8; cidx += 256) {
        int s  = cidx >> 3;
        int d0 = (cidx & 7) * 8;
        f16 kv[8], vv[8];
        if (s < TT) {
#pragma unroll
            for (int j = 0; j < 8; ++j) {
                kv[j] = Kg[(long)s * QKVW + d0 + j];
                vv[j] = Vg[(long)s * QKVW + d0 + j];
            }
        } else {
#pragma unroll
            for (int j = 0; j < 8; ++j) { kv[j] = (f16)0.0f; vv[j] = (f16)0.0f; }
        }
#pragma unroll
        for (int j = 0; j < 8; ++j) {
            Ks[s * HD + d0 + j] = kv[j];
            Vs[s * HD + d0 + j] = vv[j];
        }
    }
    __syncthreads();

    f16* Pw = Ps + wid * 16 * 32;
    const int n = lane & 15;
    const float scale = 0.125f;          // 1/sqrt(64)

    // Each wave owns query tiles wid, wid+8, wid+16, wid+24 (32 tiles cover 512 >= 500 rows)
    for (int qt = wid; qt < STILES; qt += 8) {
        // Q fragments (A layout), rows >= T zero-padded
        const int  m  = lane & 15;
        const int  t0 = qt * 16 + m;
        const bool tv = (t0 < TT);
        v16h qa0, qa1;
#pragma unroll
        for (int e = 0; e < 16; ++e) {
            int k = amap_k(e, lane);
            qa0[e] = tv ? Qg[(long)t0 * QKVW + k]      : (f16)0.0f;
            qa1[e] = tv ? Qg[(long)t0 * QKVW + 32 + k] : (f16)0.0f;
        }

        float mrow[8], lrow[8];
        v8f acc[4];
#pragma unroll
        for (int i = 0; i < 8; ++i) { mrow[i] = -1e30f; lrow[i] = 0.0f; }
#pragma unroll
        for (int nc = 0; nc < 4; ++nc) acc[nc] = (v8f){};

        for (int st = 0; st < STILES; st += 2) {
            // ---- scores for key tiles st, st+1 : S = Q (16x64) @ K^T ----
            const int col0 = st * 16 + n;
            const int col1 = col0 + 16;
            v16h kb0a, kb0b, kb1a, kb1b;   // B frags: Ks column-major gather
#pragma unroll
            for (int e = 0; e < 16; ++e) {
                int kk = bmap_k(e, lane);
                kb0a[e] = Ks[col0 * HD + kk];        // head dims 0..31
                kb0b[e] = Ks[col0 * HD + 32 + kk];   // head dims 32..63
                kb1a[e] = Ks[col1 * HD + kk];
                kb1b[e] = Ks[col1 * HD + 32 + kk];
            }
            v8f s0 = {}, s1 = {};
            s0 = __builtin_amdgcn_wmma_f32_16x16x32_f16(false, qa0, false, kb0a, (short)0, s0, false, false);
            s0 = __builtin_amdgcn_wmma_f32_16x16x32_f16(false, qa1, false, kb0b, (short)0, s0, false, false);
            s1 = __builtin_amdgcn_wmma_f32_16x16x32_f16(false, qa0, false, kb1a, (short)0, s1, false, false);
            s1 = __builtin_amdgcn_wmma_f32_16x16x32_f16(false, qa1, false, kb1b, (short)0, s1, false, false);

            const bool mask0 = (col0 >= TT);
            const bool mask1 = (col1 >= TT);

            // ---- online softmax update (per C-layout row i) ----
#pragma unroll
            for (int i = 0; i < 8; ++i) {
                float v0 = mask0 ? -1e30f : s0[i] * scale;
                float v1 = mask1 ? -1e30f : s1[i] * scale;
                float tm = fmaxf(v0, v1);
                tm = fmaxf(tm, __shfl_xor(tm, 1, 32));
                tm = fmaxf(tm, __shfl_xor(tm, 2, 32));
                tm = fmaxf(tm, __shfl_xor(tm, 4, 32));
                tm = fmaxf(tm, __shfl_xor(tm, 8, 32));
                float mnew = fmaxf(mrow[i], tm);
                float corr = __expf(mrow[i] - mnew);
                float p0 = __expf(v0 - mnew);
                float p1 = __expf(v1 - mnew);
                float ps = p0 + p1;
                ps += __shfl_xor(ps, 1, 32);
                ps += __shfl_xor(ps, 2, 32);
                ps += __shfl_xor(ps, 4, 32);
                ps += __shfl_xor(ps, 8, 32);
                lrow[i] = lrow[i] * corr + ps;
                mrow[i] = mnew;
#pragma unroll
                for (int nc = 0; nc < 4; ++nc) acc[nc][i] *= corr;
                // stash P tile (16x32, f16) in per-wave LDS scratch for layout conversion
                int mm = i + ((lane & 16) >> 1);
                Pw[mm * 32 + n]      = (f16)p0;
                Pw[mm * 32 + 16 + n] = (f16)p1;
            }

            // ---- P (A layout) @ V (B layout), K = 32 = two s-tiles ----
            v16h pa;
#pragma unroll
            for (int e = 0; e < 16; ++e)
                pa[e] = Pw[(lane & 15) * 32 + amap_k(e, lane)];
#pragma unroll
            for (int nc = 0; nc < 4; ++nc) {
                v16h vb;
#pragma unroll
                for (int e = 0; e < 16; ++e)
                    vb[e] = Vs[(st * 16 + bmap_k(e, lane)) * HD + nc * 16 + n];
                acc[nc] = __builtin_amdgcn_wmma_f32_16x16x32_f16(false, pa, false, vb,
                                                                 (short)0, acc[nc], false, false);
            }
        }

        // ---- finalize: y[b][t][h*64 + d] = acc / l ----
#pragma unroll
        for (int i = 0; i < 8; ++i) {
            float inv = 1.0f / lrow[i];
            int   mm  = i + ((lane & 16) >> 1);
            int   t   = qt * 16 + mm;
            if (t < TT) {
#pragma unroll
                for (int nc = 0; nc < 4; ++nc) {
                    y[((long)b * TT + t) * NE + h * HD + nc * 16 + n] = (f16)(acc[nc][i] * inv);
                }
            }
        }
    }
}

// =================== Kernel 3: output projection ===================
// out[r][n] = sum_k y[r][k] * w_proj[n][k] + b_proj[n]   (f32 out)
// One wave per 16x64 strip, same structure as kernel 1.
__global__ __launch_bounds__(128)
void out_proj_kernel(const f16* __restrict__ yin,
                     const f16* __restrict__ wh,
                     const float* __restrict__ bias,
                     float* __restrict__ out)
{
    const int lane = threadIdx.x & 31;
    const int wid  = threadIdx.x >> 5;
    const int mt   = blockIdx.x * 4 + wid;    // grid.x = 500
    const int ntb  = blockIdx.y;              // grid.y = 8 (4 N-tiles each)
    const long row = (long)mt * 16 + (lane & 15);
    const int  n   = lane & 15;

    v8f c[4];
#pragma unroll
    for (int j = 0; j < 4; ++j) c[j] = (v8f){};

#pragma unroll
    for (int kb = 0; kb < NE; kb += 32) {
        v16h a;
#pragma unroll
        for (int e = 0; e < 16; ++e)
            a[e] = yin[row * NE + kb + amap_k(e, lane)];
#pragma unroll
        for (int j = 0; j < 4; ++j) {
            const long ncol = (long)(ntb * 4 + j) * 16 + n;
            v16h b;
#pragma unroll
            for (int e = 0; e < 16; ++e)
                b[e] = wh[ncol * NE + kb + bmap_k(e, lane)];
            c[j] = __builtin_amdgcn_wmma_f32_16x16x32_f16(false, a, false, b,
                                                          (short)0, c[j], false, false);
        }
    }
#pragma unroll
    for (int j = 0; j < 4; ++j) {
        const long ncol = (long)(ntb * 4 + j) * 16 + n;
        const float bn  = bias[ncol];
#pragma unroll
        for (int i = 0; i < 8; ++i) {
            int  mm = i + ((lane & 16) >> 1);
            long r  = (long)mt * 16 + mm;
            out[r * NE + ncol] = c[j][i] + bn;
        }
    }
}

extern "C" void kernel_launch(void* const* d_in, const int* in_sizes, int n_in,
                              void* d_out, int out_size, void* d_ws, size_t ws_size,
                              hipStream_t stream) {
    const float* x      = (const float*)d_in[0];
    const float* w_attn = (const float*)d_in[1];
    const float* b_attn = (const float*)d_in[2];
    const float* w_proj = (const float*)d_in[3];
    const float* b_proj = (const float*)d_in[4];
    float* out = (float*)d_out;

    // Workspace layout (f16):
    //   x_h   [64*500*256]   = 8.192M halves
    //   wa_h  [1536*256]     = 0.393M
    //   wp_h  [512*512]      = 0.262M
    //   qkv   [64*500*1536]  = 49.152M  (~98.3MB)
    //   y     [64*500*512]   = 16.384M  (~32.8MB)
    const long n_x  = (long)BB * TT * INC;
    const long n_wa = (long)QKVW * INC;
    const long n_wp = (long)NE * NE;
    f16* x_h    = (f16*)d_ws;
    f16* wa_h   = x_h  + n_x;
    f16* wp_h   = wa_h + n_wa;
    f16* qkv_ws = wp_h + n_wp;
    f16* y_ws   = qkv_ws + (long)BB * TT * QKVW;

    // 0) One-shot f32->f16 conversion of x and weights (halves GEMM input traffic)
    cvt_kernel<<<dim3((unsigned)(n_x  / 1024)), 256, 0, stream>>>(x,      x_h,  n_x);
    cvt_kernel<<<dim3((unsigned)(n_wa / 1024)), 256, 0, stream>>>(w_attn, wa_h, n_wa);
    cvt_kernel<<<dim3((unsigned)(n_wp / 1024)), 256, 0, stream>>>(w_proj, wp_h, n_wp);

    // 1) QKV projection: 500 x 24 blocks, 4 waves each, wave = 16x64 strip
    qkv_proj_kernel<<<dim3(500, QKVW / 64), 128, 0, stream>>>(x_h, wa_h, b_attn, qkv_ws);

    // 2) Fused flash attention: one workgroup per (b, h); K+V+P in LDS (136KB < 320KB/WGP)
    size_t smem = (size_t)(2 * 512 * HD + 8 * 16 * 32) * sizeof(f16);
    attn_kernel<<<dim3(BB * NH), 256, smem, stream>>>(qkv_ws, y_ws);

    // 3) Output projection: 500 x 8 blocks, 4 waves each, wave = 16x64 strip
    out_proj_kernel<<<dim3(500, NE / 64), 128, 0, stream>>>(y_ws, wp_h, b_proj, out);
}
